// ReceptorEncoderGVP_66898410602532
// MI455X (gfx1250) — matrix operand order repaired and checked by reading, not compiled, not run on hardware
//
#include <hip/hip_runtime.h>
#include <math.h>

typedef __attribute__((ext_vector_type(16))) _Float16 v16h;
typedef __attribute__((ext_vector_type(8)))  _Float16 v8h;
typedef __attribute__((ext_vector_type(8)))  float    v8f;

static constexpr int   kN    = 25000;
static constexpr int   kE    = 800000;
static constexpr float kEps  = 1e-8f;
static constexpr float kZinv = 0.1f;   // 1/Z, Z = 10

// fast device math (eps-guarded uses, so native rcp/rsq/sqrt are fine)
__device__ __forceinline__ float frcp(float x)  { return __builtin_amdgcn_rcpf(x); }
__device__ __forceinline__ float frsq(float x)  { return __builtin_amdgcn_rsqf(x); }
__device__ __forceinline__ float fsqrt(float x) { return __builtin_amdgcn_sqrtf(x); }
__device__ __forceinline__ float silu_f(float x) { return x * frcp(1.0f + __expf(-x)); }
__device__ __forceinline__ float pair_sum(float x) { return x + __shfl_xor(x, 16, 32); }

// ---- CDNA5 wave32 WMMA f16 16x16x32 fragment loaders ----
// A (16M x 32K, f16): lane<16 -> row=lane,    halves K[0..7] then K[16..23]
//                     lane>=16 -> row=lane-16, halves K[8..15] then K[24..31]
__device__ __forceinline__ v16h ld_afrag(const _Float16* a, int lda) {
  const int lane = threadIdx.x & 31;
  const _Float16* p = a + (lane & 15) * lda + ((lane >> 4) << 3);
  v8h lo = *(const v8h*)p;
  v8h hi = *(const v8h*)(p + 16);
  v16h r;
#pragma unroll
  for (int i = 0; i < 8; ++i) { r[i] = lo[i]; r[i + 8] = hi[i]; }
  return r;
}

// B (32K x 16N, f16) stored N-major (Bt[n][k], ldb = padded K):
// lane<16 -> col=lane, K[0..7] then K[16..23]; lane>=16 -> col=lane-16, K[8..15] then K[24..31]
__device__ __forceinline__ v16h ld_bfrag(const _Float16* bt, int ldb) {
  const int lane = threadIdx.x & 31;
  const _Float16* p = bt + (lane & 15) * ldb + ((lane >> 4) << 3);
  v8h lo = *(const v8h*)p;
  v8h hi = *(const v8h*)(p + 16);
  v16h r;
#pragma unroll
  for (int i = 0; i < 8; ++i) { r[i] = lo[i]; r[i + 8] = hi[i]; }
  return r;
}

template <int KT>
__device__ __forceinline__ v8f wmma_mm(const _Float16* a, int lda,
                                       const _Float16* bt, int ldb) {
  v8f acc = {};
#pragma unroll
  for (int kt = 0; kt < KT; ++kt) {
    v16h af = ld_afrag(a + kt * 32, lda);
    v16h bf = ld_bfrag(bt + kt * 32, ldb);
    acc = __builtin_amdgcn_wmma_f32_16x16x32_f16(false, af, false, bf,
                                                 (short)0, acc, false, false);
  }
  return acc;
}

// Convert f32 weight [K,N] row-major -> f16 transposed Bt[n*Kp + k], zero-padded K.
__global__ void convert_wt_kernel(const float* __restrict__ W, _Float16* __restrict__ Wt,
                                  int K, int N, int Kp) {
  const int i = blockIdx.x * blockDim.x + threadIdx.x;
  if (i >= N * Kp) return;
  const int n = i / Kp, k = i - n * Kp;
  Wt[i] = (k < K) ? (_Float16)W[(size_t)k * N + n] : (_Float16)0.0f;
}

// s = LN(silu(silu(h@W1+b1)@W2+b2))
__global__ __launch_bounds__(64) void embed_kernel(
    const float* __restrict__ h, const _Float16* __restrict__ W1T, const float* __restrict__ b1,
    const _Float16* __restrict__ W2T, const float* __restrict__ b2,
    const float* __restrict__ ng, const float* __restrict__ nb,
    float* __restrict__ sbuf) {
  __shared__ __align__(32) _Float16 A[32][64];
  __shared__ __align__(32) _Float16 A2[32][128];
  __shared__ __align__(32) _Float16 A3[32][128];
  const int tid = threadIdx.x;
  const int lane = tid & 31, wave = tid >> 5;
  const int row = wave * 16 + (lane & 15);
  const int half = lane >> 4;
  const int n0 = blockIdx.x * 32;
  const int node = n0 + row;

  if (node < kN) {
    const float* p = h + (size_t)node * 64 + half * 32;
    _Float16* ar = &A[row][half * 32];
#pragma unroll 8
    for (int j = 0; j < 32; ++j) ar[j] = (_Float16)p[j];
  } else {
    _Float16* ar = &A[row][half * 32];
    for (int j = 0; j < 32; ++j) ar[j] = (_Float16)0.0f;
  }
  __syncthreads();
  const int n_lo = lane & 15;
  const int m_of = half * 8 + wave * 16;
#pragma unroll
  for (int nt = 0; nt < 8; ++nt) {
    v8f acc = wmma_mm<2>(&A[wave * 16][0], 64, W1T + nt * 16 * 64, 64);
    const int n = nt * 16 + n_lo;
    const float b = b1[n];
#pragma unroll
    for (int r = 0; r < 8; ++r) A2[m_of + r][n] = (_Float16)silu_f(acc[r] + b);
  }
  __syncthreads();
#pragma unroll
  for (int nt = 0; nt < 8; ++nt) {
    v8f acc = wmma_mm<4>(&A2[wave * 16][0], 128, W2T + nt * 16 * 128, 128);
    const int n = nt * 16 + n_lo;
    const float b = b2[n];
#pragma unroll
    for (int r = 0; r < 8; ++r) A3[m_of + r][n] = (_Float16)silu_f(acc[r] + b);
  }
  __syncthreads();
  if (node < kN) {
    const _Float16* fr = &A3[row][half * 64];
    float pmu = 0.f;
    for (int j = 0; j < 64; ++j) pmu += (float)fr[j];
    const float mu = pair_sum(pmu) * (1.0f / 128.0f);
    float pvar = 0.f;
    for (int j = 0; j < 64; ++j) { const float d = (float)fr[j] - mu; pvar += d * d; }
    const float var = pair_sum(pvar) * (1.0f / 128.0f);
    const float inv = frsq(var + 1e-5f);
    float* so = sbuf + (size_t)node * 128 + half * 64;
    for (int j = 0; j < 64; ++j) {
      const int jj = half * 64 + j;
      so[j] = ((float)fr[j] - mu) * inv * ng[jj] + nb[jj];
    }
  }
}

// Per-edge GVP message + atomic scatter-add of sm/Z, vm/Z.
__global__ __launch_bounds__(64) void edge_msg_kernel(
    const float* __restrict__ sbuf, const float* __restrict__ vbuf,
    const float* __restrict__ coords,
    const int* __restrict__ esrc, const int* __restrict__ edst,
    const float* __restrict__ Wh, const float* __restrict__ Wu,
    const _Float16* __restrict__ WsT, const float* __restrict__ bs,
    const _Float16* __restrict__ WgT, const float* __restrict__ bg,
    const float* __restrict__ lng, const float* __restrict__ lnb,
    float* __restrict__ sacc, float* __restrict__ vacc) {
  __shared__ __align__(32) _Float16 A[32][160];
  __shared__ __align__(32) _Float16 F[32][128];
  __shared__ float VU[32][48];
  __shared__ float G[32][16];
  const int tid = threadIdx.x;
  const int lane = tid & 31, wave = tid >> 5;
  const int row = wave * 16 + (lane & 15);
  const int half = lane >> 4;
  const int e0 = blockIdx.x * 32;
  const int e = e0 + row;

  // ---- phase 1: gather + vector einsum (lane pair per edge) ----
  if (e < kE) {
    const int src = esrc[e], dst = edst[e];
    const float* sp = sbuf + (size_t)src * 128 + half * 64;
    _Float16* ar = &A[row][half * 64];
#pragma unroll 8
    for (int j = 0; j < 64; ++j) ar[j] = (_Float16)sp[j];
    float ve[17][3];
#pragma unroll
    for (int c = 0; c < 3; ++c)
      ve[0][c] = coords[src * 3 + c] - coords[dst * 3 + c];
    const float* vp = vbuf + (size_t)src * 48;
    for (int u = 0; u < 16; ++u)
#pragma unroll
      for (int c = 0; c < 3; ++c) ve[u + 1][c] = vp[u * 3 + c];
    float vh[17][3];   // duplicated across the pair; outputs split below
    for (int hh = 0; hh < 17; ++hh) vh[hh][0] = vh[hh][1] = vh[hh][2] = 0.f;
    for (int v = 0; v < 17; ++v)
      for (int hh = 0; hh < 17; ++hh) {
        const float w = Wh[v * 17 + hh];
        vh[hh][0] += ve[v][0] * w; vh[hh][1] += ve[v][1] * w; vh[hh][2] += ve[v][2] * w;
      }
    const int h_lo = half ? 8 : 0, h_hi = half ? 17 : 8;
    for (int hh = h_lo; hh < h_hi; ++hh) {
      const float d2 = vh[hh][0]*vh[hh][0] + vh[hh][1]*vh[hh][1] + vh[hh][2]*vh[hh][2];
      A[row][128 + hh] = (_Float16)fsqrt(d2 + kEps);
    }
    if (half)
      for (int j = 145; j < 160; ++j) A[row][j] = (_Float16)0.0f;
    for (int u = half * 8; u < half * 8 + 8; ++u) {
      float a0 = 0.f, a1 = 0.f, a2 = 0.f;
      for (int hh = 0; hh < 17; ++hh) {
        const float w = Wu[hh * 16 + u];
        a0 += vh[hh][0] * w; a1 += vh[hh][1] * w; a2 += vh[hh][2] * w;
      }
      VU[row][u * 3 + 0] = a0; VU[row][u * 3 + 1] = a1; VU[row][u * 3 + 2] = a2;
    }
  } else {
    _Float16* ar = &A[row][half * 80];
    for (int j = 0; j < 80; ++j) ar[j] = (_Float16)0.0f;
    for (int k = 0; k < 24; ++k) VU[row][half * 24 + k] = 0.f;
  }
  __syncthreads();

  // ---- phase 2: feats = silu(A @ WsT + bs)  (f16 WMMA, f32 acc) ----
  const int n_lo = lane & 15;
  const int m_of = half * 8 + wave * 16;
#pragma unroll
  for (int nt = 0; nt < 8; ++nt) {
    v8f acc = wmma_mm<5>(&A[wave * 16][0], 160, WsT + nt * 16 * 160, 160);
    const int n = nt * 16 + n_lo;
    const float b = bs[n];
#pragma unroll
    for (int r = 0; r < 8; ++r) F[m_of + r][n] = (_Float16)silu_f(acc[r] + b);
  }
  __syncthreads();

  // ---- phase 3: gate = silu(F @ WgT + bg) ----
  {
    v8f acc = wmma_mm<4>(&F[wave * 16][0], 128, WgT, 128);
    const float b = bg[n_lo];
#pragma unroll
    for (int r = 0; r < 8; ++r) G[m_of + r][n_lo] = silu_f(acc[r] + b);
  }
  __syncthreads();

  // ---- phase 4: LN + vec-RMS + atomic scatter (lane pair per edge) ----
  if (e < kE) {
    const int dst = edst[e];
    const _Float16* fr = &F[row][half * 64];
    float pmu = 0.f;
    for (int j = 0; j < 64; ++j) pmu += (float)fr[j];
    const float mu = pair_sum(pmu) * (1.0f / 128.0f);
    float pvar = 0.f;
    for (int j = 0; j < 64; ++j) { const float d = (float)fr[j] - mu; pvar += d * d; }
    const float var = pair_sum(pvar) * (1.0f / 128.0f);
    const float inv = frsq(var + 1e-5f);
    float* so = sacc + (size_t)dst * 128 + half * 64;
    for (int j = 0; j < 64; ++j) {
      const int jj = half * 64 + j;
      const float sm = ((float)fr[j] - mu) * inv * lng[jj] + lnb[jj];
      atomicAdd(&so[j], sm * kZinv);
    }
    float vo[24];
    float pn2 = 0.f;
    for (int u = 0; u < 8; ++u) {
      const int uu = half * 8 + u;
      const float g = G[row][uu];
#pragma unroll
      for (int c = 0; c < 3; ++c) {
        const float x = VU[row][uu * 3 + c] * g;
        vo[u * 3 + c] = x; pn2 += x * x;
      }
    }
    const float n2 = pair_sum(pn2);
    const float ri = frsq(n2 * (1.0f / 16.0f) + kEps);
    float* vv = vacc + (size_t)dst * 48 + half * 24;
    for (int k = 0; k < 24; ++k) atomicAdd(&vv[k], vo[k] * ri * kZinv);
  }
}

// s1 = LN(s+msg); v1 = vecLN(v+msg); (sr,vr) = GVP(s1,v1); out = LN(s1+sr), vecLN(v1+vr)
__global__ __launch_bounds__(64) void node_upd_kernel(
    const float* __restrict__ sacc, const float* __restrict__ vacc,
    const float* __restrict__ Wh, const float* __restrict__ Wu,
    const _Float16* __restrict__ WsT, const float* __restrict__ bs,
    const _Float16* __restrict__ WgT, const float* __restrict__ bg,
    const float* __restrict__ nlg, const float* __restrict__ nlb,
    const float* __restrict__ lg, const float* __restrict__ lb,
    float* __restrict__ sout, float* __restrict__ vout) {
  __shared__ __align__(32) _Float16 A[32][160];
  __shared__ __align__(32) _Float16 F[32][128];
  __shared__ float S1[32][128];
  __shared__ float V1[32][48];
  __shared__ float VU[32][48];
  __shared__ float G[32][16];
  const int tid = threadIdx.x;
  const int lane = tid & 31, wave = tid >> 5;
  const int row = wave * 16 + (lane & 15);
  const int half = lane >> 4;
  const int n0 = blockIdx.x * 32;
  const int node = n0 + row;

  // ---- phase 1a: LN(s_acc) -> S1/A ; vecLN(v_acc) -> V1 ----
  if (node < kN) {
    const float* sp = sacc + (size_t)node * 128 + half * 64;
    float pmu = 0.f;
    for (int j = 0; j < 64; ++j) pmu += sp[j];
    const float mu = pair_sum(pmu) * (1.0f / 128.0f);
    float pvar = 0.f;
    for (int j = 0; j < 64; ++j) { const float d = sp[j] - mu; pvar += d * d; }
    const float var = pair_sum(pvar) * (1.0f / 128.0f);
    const float inv = frsq(var + 1e-5f);
    for (int j = 0; j < 64; ++j) {
      const int jj = half * 64 + j;
      const float s1 = (sp[j] - mu) * inv * lg[jj] + lb[jj];
      S1[row][jj] = s1;
      A[row][jj] = (_Float16)s1;
    }
    const float* vp = vacc + (size_t)node * 48 + half * 24;
    float pn2 = 0.f;
    for (int k = 0; k < 24; ++k) pn2 += vp[k] * vp[k];
    const float n2 = pair_sum(pn2);
    const float ri = frsq(n2 * (1.0f / 16.0f) + kEps);
    for (int k = 0; k < 24; ++k) V1[row][half * 24 + k] = vp[k] * ri;
  } else {
    for (int j = 0; j < 64; ++j) {
      const int jj = half * 64 + j;
      A[row][jj] = (_Float16)0.0f;
      S1[row][jj] = 0.f;
    }
    for (int k = 0; k < 24; ++k) V1[row][half * 24 + k] = 0.f;
  }
  __syncthreads();

  // ---- phase 1b: vh/sh/vu from V1 (duplicated vh, split outputs) ----
  if (node < kN) {
    float vh[16][3];
    for (int hh = 0; hh < 16; ++hh) vh[hh][0] = vh[hh][1] = vh[hh][2] = 0.f;
    for (int v = 0; v < 16; ++v)
      for (int hh = 0; hh < 16; ++hh) {
        const float w = Wh[v * 16 + hh];
        vh[hh][0] += V1[row][v * 3 + 0] * w;
        vh[hh][1] += V1[row][v * 3 + 1] * w;
        vh[hh][2] += V1[row][v * 3 + 2] * w;
      }
    for (int hh = half * 8; hh < half * 8 + 8; ++hh) {
      const float d2 = vh[hh][0]*vh[hh][0] + vh[hh][1]*vh[hh][1] + vh[hh][2]*vh[hh][2];
      A[row][128 + hh] = (_Float16)fsqrt(d2 + kEps);
    }
    if (half)
      for (int j = 144; j < 160; ++j) A[row][j] = (_Float16)0.0f;
    for (int u = half * 8; u < half * 8 + 8; ++u) {
      float a0 = 0.f, a1 = 0.f, a2 = 0.f;
      for (int hh = 0; hh < 16; ++hh) {
        const float w = Wu[hh * 16 + u];
        a0 += vh[hh][0] * w; a1 += vh[hh][1] * w; a2 += vh[hh][2] * w;
      }
      VU[row][u * 3 + 0] = a0; VU[row][u * 3 + 1] = a1; VU[row][u * 3 + 2] = a2;
    }
  } else {
    for (int j = 0; j < 16; ++j) A[row][128 + half * 16 + j] = (_Float16)0.0f;
    for (int k = 0; k < 24; ++k) VU[row][half * 24 + k] = 0.f;
  }
  __syncthreads();

  // ---- phase 2: feats WMMA ----
  const int n_lo = lane & 15;
  const int m_of = half * 8 + wave * 16;
#pragma unroll
  for (int nt = 0; nt < 8; ++nt) {
    v8f acc = wmma_mm<5>(&A[wave * 16][0], 160, WsT + nt * 16 * 160, 160);
    const int n = nt * 16 + n_lo;
    const float b = bs[n];
#pragma unroll
    for (int r = 0; r < 8; ++r) F[m_of + r][n] = (_Float16)silu_f(acc[r] + b);
  }
  __syncthreads();

  // ---- phase 3: gate WMMA ----
  {
    v8f acc = wmma_mm<4>(&F[wave * 16][0], 128, WgT, 128);
    const float b = bg[n_lo];
#pragma unroll
    for (int r = 0; r < 8; ++r) G[m_of + r][n_lo] = silu_f(acc[r] + b);
  }
  __syncthreads();

  // ---- phase 4: residuals + double LN + vec-RMS, write outputs ----
  if (node < kN) {
    const _Float16* fr = &F[row][half * 64];
    float pmu = 0.f;
    for (int j = 0; j < 64; ++j) pmu += (float)fr[j];
    const float mu = pair_sum(pmu) * (1.0f / 128.0f);
    float pvar = 0.f;
    for (int j = 0; j < 64; ++j) { const float d = (float)fr[j] - mu; pvar += d * d; }
    const float var = pair_sum(pvar) * (1.0f / 128.0f);
    const float inv = frsq(var + 1e-5f);
    float s2h[64];
    float pmu2 = 0.f;
    for (int j = 0; j < 64; ++j) {
      const int jj = half * 64 + j;
      const float sr = ((float)fr[j] - mu) * inv * nlg[jj] + nlb[jj];
      s2h[j] = S1[row][jj] + sr;
      pmu2 += s2h[j];
    }
    const float mu2 = pair_sum(pmu2) * (1.0f / 128.0f);
    float pvar2 = 0.f;
    for (int j = 0; j < 64; ++j) { const float d = s2h[j] - mu2; pvar2 += d * d; }
    const float var2 = pair_sum(pvar2) * (1.0f / 128.0f);
    const float inv2 = frsq(var2 + 1e-5f);
    float* so = sout + (size_t)node * 128 + half * 64;
    for (int j = 0; j < 64; ++j) {
      const int jj = half * 64 + j;
      so[j] = (s2h[j] - mu2) * inv2 * lg[jj] + lb[jj];
    }
    float voh[24];
    float pna = 0.f;
    for (int u = 0; u < 8; ++u) {
      const int uu = half * 8 + u;
      const float g = G[row][uu];
#pragma unroll
      for (int c = 0; c < 3; ++c) {
        const float x = VU[row][uu * 3 + c] * g;
        voh[u * 3 + c] = x; pna += x * x;
      }
    }
    const float na = pair_sum(pna);
    const float ra = frsq(na * (1.0f / 16.0f) + kEps);
    float pnb = 0.f;
    for (int k = 0; k < 24; ++k) {
      const float y = V1[row][half * 24 + k] + voh[k] * ra;
      voh[k] = y; pnb += y * y;
    }
    const float nb2 = pair_sum(pnb);
    const float rb = frsq(nb2 * (1.0f / 16.0f) + kEps);
    float* vo = vout + (size_t)node * 48 + half * 24;
    for (int k = 0; k < 24; ++k) vo[k] = voh[k] * rb;
  }
}

extern "C" void kernel_launch(void* const* d_in, const int* in_sizes, int n_in,
                              void* d_out, int out_size, void* d_ws, size_t ws_size,
                              hipStream_t stream) {
  (void)in_sizes; (void)n_in; (void)out_size; (void)ws_size;
  const float* node_h = (const float*)d_in[0];
  const float* coords = (const float*)d_in[1];
  const int*   esrc   = (const int*)d_in[2];
  const int*   edst   = (const int*)d_in[3];
  const float* emb_W1 = (const float*)d_in[4];
  const float* emb_b1 = (const float*)d_in[5];
  const float* emb_W2 = (const float*)d_in[6];
  const float* emb_b2 = (const float*)d_in[7];
  const float* norm_g = (const float*)d_in[8];
  const float* norm_b = (const float*)d_in[9];
  const float* em_Wh  = (const float*)d_in[10];
  const float* em_Wu  = (const float*)d_in[11];
  const float* em_Ws  = (const float*)d_in[12];
  const float* em_bs  = (const float*)d_in[13];
  const float* em_Wg  = (const float*)d_in[14];
  const float* em_bg  = (const float*)d_in[15];
  const float* em_lng = (const float*)d_in[16];
  const float* em_lnb = (const float*)d_in[17];
  const float* nu_Wh  = (const float*)d_in[18];
  const float* nu_Wu  = (const float*)d_in[19];
  const float* nu_Ws  = (const float*)d_in[20];
  const float* nu_bs  = (const float*)d_in[21];
  const float* nu_Wg  = (const float*)d_in[22];
  const float* nu_bg  = (const float*)d_in[23];
  const float* nu_lng = (const float*)d_in[24];
  const float* nu_lnb = (const float*)d_in[25];
  const float* ln_g   = (const float*)d_in[26];
  const float* ln_b   = (const float*)d_in[27];

  char* ws = (char*)d_ws;
  size_t off = 0;
  auto alloc = [&](size_t bytes) -> char* {
    char* p = ws + off;
    off = (off + bytes + 255) & ~(size_t)255;
    return p;
  };
  float* sbuf = (float*)alloc((size_t)kN * 128 * 4);
  float* vbuf = (float*)alloc((size_t)kN * 48 * 4);
  float* sacc = (float*)alloc((size_t)kN * 128 * 4);
  float* vacc = (float*)alloc((size_t)kN * 48 * 4);
  _Float16* w1t   = (_Float16*)alloc((size_t)128 * 64 * 2);
  _Float16* w2t   = (_Float16*)alloc((size_t)128 * 128 * 2);
  _Float16* emwst = (_Float16*)alloc((size_t)3 * 128 * 160 * 2);
  _Float16* emwgt = (_Float16*)alloc((size_t)3 * 16 * 128 * 2);
  _Float16* nuwst = (_Float16*)alloc((size_t)3 * 128 * 160 * 2);
  _Float16* nuwgt = (_Float16*)alloc((size_t)3 * 16 * 128 * 2);

  auto conv = [&](const float* W, _Float16* Wt, int K, int Nc, int Kp) {
    const int tot = Nc * Kp;
    convert_wt_kernel<<<(tot + 255) / 256, 256, 0, stream>>>(W, Wt, K, Nc, Kp);
  };
  conv(emb_W1, w1t, 64, 128, 64);
  conv(emb_W2, w2t, 128, 128, 128);
  for (int l = 0; l < 3; ++l) {
    conv(em_Ws + (size_t)l * 145 * 128, emwst + (size_t)l * 128 * 160, 145, 128, 160);
    conv(em_Wg + (size_t)l * 128 * 16,  emwgt + (size_t)l * 16 * 128, 128, 16, 128);
    conv(nu_Ws + (size_t)l * 144 * 128, nuwst + (size_t)l * 128 * 160, 144, 128, 160);
    conv(nu_Wg + (size_t)l * 128 * 16,  nuwgt + (size_t)l * 16 * 128, 128, 16, 128);
  }

  embed_kernel<<<(kN + 31) / 32, 64, 0, stream>>>(
      node_h, w1t, emb_b1, w2t, emb_b2, norm_g, norm_b, sbuf);
  hipMemsetAsync(vbuf, 0, (size_t)kN * 48 * 4, stream);

  float* s_final = (float*)d_out;
  float* v_final = (float*)d_out + (size_t)kN * 128;

  for (int l = 0; l < 3; ++l) {
    hipMemcpyAsync(sacc, sbuf, (size_t)kN * 128 * 4, hipMemcpyDeviceToDevice, stream);
    hipMemcpyAsync(vacc, vbuf, (size_t)kN * 48 * 4, hipMemcpyDeviceToDevice, stream);
    edge_msg_kernel<<<kE / 32, 64, 0, stream>>>(
        sbuf, vbuf, coords, esrc, edst,
        em_Wh + (size_t)l * 289, em_Wu + (size_t)l * 272,
        emwst + (size_t)l * 128 * 160, em_bs + (size_t)l * 128,
        emwgt + (size_t)l * 16 * 128,  em_bg + (size_t)l * 16,
        em_lng + (size_t)l * 128, em_lnb + (size_t)l * 128,
        sacc, vacc);
    float* so = (l == 2) ? s_final : sbuf;
    float* vo = (l == 2) ? v_final : vbuf;
    node_upd_kernel<<<(kN + 31) / 32, 64, 0, stream>>>(
        sacc, vacc,
        nu_Wh + (size_t)l * 256, nu_Wu + (size_t)l * 256,
        nuwst + (size_t)l * 128 * 160, nu_bs + (size_t)l * 128,
        nuwgt + (size_t)l * 16 * 128,  nu_bg + (size_t)l * 16,
        nu_lng + (size_t)l * 128, nu_lnb + (size_t)l * 128,
        ln_g + (size_t)l * 128, ln_b + (size_t)l * 128,
        so, vo);
  }
}
